// FP_Module_angle_label_v2_18433999634715
// MI455X (gfx1250) — compile-verified, matrix operand order checked by reading, not run on hardware
//
#include <hip/hip_runtime.h>
#include <hip/hip_bf16.h>

// ---------------------------------------------------------------------------
// Problem constants (from the reference): B=8, N1=8192, N2=2048, C2=256, K=3
// IN_CH = 261 (padded to 288 = 9*32 for WMMA K-steps), MLP0 = MLP1 = 256.
// ---------------------------------------------------------------------------
#define BB      8
#define N1      8192
#define N2      2048
#define C2      256
#define KNN     3
#define INCH    261
#define KPAD    288           // 9 * 32
#define OUTCH   256
#define NQ      (BB * N1)     // 65536 queries
#define MROWS   (NQ * KNN)    // 196608 GEMM rows
#define MT      48            // rows per block tile (3 x 16, and 16 k-groups)
#define NTILES  (MROWS / MT)  // 4096 blocks

typedef __attribute__((ext_vector_type(16))) _Float16 v16h;
typedef __attribute__((ext_vector_type(8)))  float    v8f;

// Workspace layout (bytes)
#define WS_IDX   0                       // int[NQ*3]          = 786432
#define WS_S0    786432                  // float[256]
#define WS_T0    (WS_S0 + 1024)
#define WS_S1    (WS_T0 + 1024)
#define WS_T1    (WS_S1 + 1024)
#define WS_WP0   (WS_T1 + 1024)          // f16[288*256] = 147456 B
#define WS_WP1   (WS_WP0 + 147456)       // f16[256*256] = 131072 B

// Dynamic LDS layout for the fused kernel (bytes)
#define L_XF     0                        // f16[3*9*512]  = 27648 B in a 55296 B slot  (A-frags L0)
#define L_HF     55296                    // f16[3*8*512]  = 24576 B in a 49152 B slot  (A-frags L1)
#define L_EXT    (L_HF + 49152)           // float[48*3]   = 576 B
#define L_FB     (L_EXT + 576)            // int[48]       = 192 B
#define L_TOTAL  (L_FB + 192)             // 105216 B  (Y f32[48*256] aliases L_XF)

// ---------------------------------------------------------------------------
// Kernel: fold BatchNorm into per-channel scale/bias
// ---------------------------------------------------------------------------
__global__ __launch_bounds__(256) void bn_fold_kernel(
    const float* __restrict__ g0, const float* __restrict__ b0,
    const float* __restrict__ m0, const float* __restrict__ v0,
    const float* __restrict__ g1, const float* __restrict__ b1,
    const float* __restrict__ m1, const float* __restrict__ v1,
    float* __restrict__ s0, float* __restrict__ t0,
    float* __restrict__ s1, float* __restrict__ t1)
{
    int o = threadIdx.x;
    float sc0 = g0[o] * rsqrtf(v0[o] + 1e-5f);
    s0[o] = sc0;  t0[o] = b0[o] - m0[o] * sc0;
    float sc1 = g1[o] * rsqrtf(v1[o] + 1e-5f);
    s1[o] = sc1;  t1[o] = b1[o] - m1[o] * sc1;
}

// ---------------------------------------------------------------------------
// Weight repack: W (OUT x IN, row-major) -> B-matrix fragments (K x N), f16.
// Fragment (ks, nt) holds K=32 x N=16; lane l reads 16 contiguous f16 (32B).
// ---------------------------------------------------------------------------
__global__ __launch_bounds__(256) void pack_w_kernel(
    const float* __restrict__ W, _Float16* __restrict__ wp, int in_ch)
{
    int k = blockIdx.x;      // 0..KPAD-1 (or 0..255 for layer 1)
    int n = threadIdx.x;     // 0..255 output channel
    float val = (k < in_ch) ? W[n * in_ch + k] : 0.0f;
    int ks   = k >> 5;
    int kk   = k & 31;
    int hi   = (kk >> 3) & 1;
    int v    = ((kk & 16) >> 2) + ((kk & 7) >> 1);
    int half = kk & 1;
    int lane = (n & 15) + 16 * hi;
    wp[(((ks << 4) + (n >> 4)) << 9) + (lane << 4) + v * 2 + half] = (_Float16)val;
}

// ---------------------------------------------------------------------------
// kNN: one thread per query; xyz2 staged through LDS in 128-point tiles.
// Strict < insertion keeps lowest-index-first tie order (matches top_k).
// ---------------------------------------------------------------------------
__global__ __launch_bounds__(256) void knn_kernel(
    const float* __restrict__ xyz1, const float* __restrict__ xyz2,
    int* __restrict__ idx)
{
    __shared__ float tile[128 * 3];
    int q = blockIdx.x * 256 + threadIdx.x;
    int b = q >> 13;                       // N1 = 8192
    float qx = xyz1[q * 3 + 0];
    float qy = xyz1[q * 3 + 1];
    float qz = xyz1[q * 3 + 2];
    const float* base = xyz2 + (size_t)b * N2 * 3;
    float d0 = 3.4e38f, d1 = 3.4e38f, d2 = 3.4e38f;
    int   i0 = 0, i1 = 0, i2 = 0;
    for (int t0 = 0; t0 < N2; t0 += 128) {
        __syncthreads();
        for (int i = threadIdx.x; i < 128 * 3; i += 256)
            tile[i] = base[t0 * 3 + i];
        __syncthreads();
        #pragma unroll 4
        for (int jj = 0; jj < 128; ++jj) {
            float dx = qx - tile[jj * 3 + 0];
            float dy = qy - tile[jj * 3 + 1];
            float dz = qz - tile[jj * 3 + 2];
            float d = dx * dx + dy * dy + dz * dz;
            int j = t0 + jj;
            if (d < d0)      { d2 = d1; i2 = i1; d1 = d0; i1 = i0; d0 = d; i0 = j; }
            else if (d < d1) { d2 = d1; i2 = i1; d1 = d;  i1 = j; }
            else if (d < d2) { d2 = d;  i2 = j; }
        }
    }
    idx[q * 3 + 0] = i0;
    idx[q * 3 + 1] = i1;
    idx[q * 3 + 2] = i2;
}

// ---------------------------------------------------------------------------
// Fused: gather features -> layer0 WMMA GEMM -> BN+ReLU -> layer1 WMMA GEMM
//        -> BN+ReLU -> sum over k=3 -> d_out.
// Block = 256 threads (8 wave32), tile = 48 rows x 256 cols.
// Tile ownership: wave w owns output tiles (m, n) for m=0..2 (compile-time)
// and n in {2w, 2w+1}.  acc[nn*3+m].  No runtime A-fragment selects.
// ---------------------------------------------------------------------------
__global__ __launch_bounds__(256) void fused_mlp_kernel(
    const int*   __restrict__ idx,
    const float* __restrict__ points1,
    const float* __restrict__ xyz2,
    const float* __restrict__ dir1,
    const float* __restrict__ feat2,
    const _Float16* __restrict__ wp0,
    const _Float16* __restrict__ wp1,
    const float* __restrict__ s0, const float* __restrict__ t0,
    const float* __restrict__ s1, const float* __restrict__ t1,
    float* __restrict__ out)
{
    extern __shared__ __align__(128) char smem[];
    _Float16* Xf   = (_Float16*)(smem + L_XF);   // layer-0 A fragments
    _Float16* Hf   = (_Float16*)(smem + L_HF);   // layer-1 A fragments
    float*    ext  = (float*)   (smem + L_EXT);  // per-row [sqd, a0, a1]
    int*      fb   = (int*)     (smem + L_FB);   // per-row feat2 base
    float*    Y    = (float*)   (smem + L_XF);   // aliases Xf (dead by then)

    const int tid  = threadIdx.x;
    const int wave = tid >> 5;
    const int lane = tid & 31;
    const int rowBase = blockIdx.x * MT;

    // ---- Phase 1: per-row geometric scalars ------------------------------
    if (tid < MT) {
        int rowg = rowBase + tid;
        int q    = rowg / 3;
        int kk   = rowg - q * 3;
        int b    = q >> 13;
        int j    = idx[q * 3 + kk];
        int pb   = b * N2 + j;
        float px = points1[q * 3 + 0], py = points1[q * 3 + 1], pz = points1[q * 3 + 2];
        float cx = xyz2[pb * 3 + 0],   cy = xyz2[pb * 3 + 1],   cz = xyz2[pb * 3 + 2];
        float rx = px - cx, ry = py - cy, rz = pz - cz;
        float sqd = rx * rx + ry * ry + rz * rz;
        float inv = 1.0f / fmaxf(sqrtf(sqd), 1e-12f);
        float fx = rx * inv, fy = ry * inv, fz = rz * inv;
        const float* d = dir1 + (size_t)pb * 6;
        ext[tid * 3 + 0] = sqd;
        ext[tid * 3 + 1] = d[0] * fx + d[1] * fy + d[2] * fz;   // angle axis 0
        ext[tid * 3 + 2] = d[3] * fx + d[4] * fy + d[5] * fz;   // angle axis 1
        fb[tid] = pb * C2;
    }
    __syncthreads();

    // ---- Phase 2: build X tile (48 x 288 f16) in A-fragment layout -------
    for (int e = tid; e < MT * KPAD; e += 256) {
        int r = e / KPAD;
        int c = e - r * KPAD;
        float val;
        if      (c < C2)                 val = feat2[fb[r] + c];
        else if (c == 256)               val = ext[r * 3 + 0];
        else if (c == 257 || c == 259)   val = ext[r * 3 + 1];
        else if (c == 258 || c == 260)   val = ext[r * 3 + 2];   // angle2 == angle1
        else                             val = 0.0f;
        int msub = r >> 4, M = r & 15;
        int ks = c >> 5, k2 = c & 31;
        int hi   = (k2 >> 3) & 1;
        int v    = ((k2 & 16) >> 2) + ((k2 & 7) >> 1);
        int half = k2 & 1;
        Xf[(msub * 9 + ks) * 512 + (M + 16 * hi) * 16 + v * 2 + half] = (_Float16)val;
    }
    __syncthreads();

    v8f acc[6];
    const v8f vzero = {0.f, 0.f, 0.f, 0.f, 0.f, 0.f, 0.f, 0.f};
    #pragma unroll
    for (int j = 0; j < 6; ++j) acc[j] = vzero;

    const int laneOff = lane << 4;        // 16 f16 per lane within a fragment

    // ---- Layer 0: 48x288 @ 288x256 ---------------------------------------
    #pragma unroll
    for (int ks = 0; ks < 9; ++ks) {
        v16h a0 = *(const v16h*)(Xf + (0 * 9 + ks) * 512 + laneOff);
        v16h a1 = *(const v16h*)(Xf + (1 * 9 + ks) * 512 + laneOff);
        v16h a2 = *(const v16h*)(Xf + (2 * 9 + ks) * 512 + laneOff);
        #pragma unroll
        for (int nn = 0; nn < 2; ++nn) {
            int n = 2 * wave + nn;
            v16h bf = *(const v16h*)(wp0 + (((ks << 4) + n) << 9) + laneOff);
            acc[nn * 3 + 0] = __builtin_amdgcn_wmma_f32_16x16x32_f16(
                false, a0, false, bf, (short)0, acc[nn * 3 + 0], false, false);
            acc[nn * 3 + 1] = __builtin_amdgcn_wmma_f32_16x16x32_f16(
                false, a1, false, bf, (short)0, acc[nn * 3 + 1], false, false);
            acc[nn * 3 + 2] = __builtin_amdgcn_wmma_f32_16x16x32_f16(
                false, a2, false, bf, (short)0, acc[nn * 3 + 2], false, false);
        }
    }

    // ---- Epilogue 0: BN + ReLU -> Hf (layer-1 A fragments) ---------------
    {
        int ln = lane & 15, lh = lane >> 4;
        int hi   = (ln >> 3) & 1;
        int half = ln & 1;
        #pragma unroll
        for (int nn = 0; nn < 2; ++nn) {
            int n   = 2 * wave + nn;
            int col = n * 16 + ln;
            float sc = s0[col], bs = t0[col];
            int ks1 = wave;                       // n >> 1
            int v   = nn * 4 + ((ln & 7) >> 1);   // (n & 1) * 4 + ...
            #pragma unroll
            for (int m = 0; m < 3; ++m) {
                #pragma unroll
                for (int r = 0; r < 8; ++r) {
                    float y = fmaxf(acc[nn * 3 + m][r] * sc + bs, 0.0f);
                    int M = r + 8 * lh;
                    Hf[(m * 8 + ks1) * 512 + (M + 16 * hi) * 16 + v * 2 + half] =
                        (_Float16)y;
                }
            }
        }
    }
    __syncthreads();   // Hf complete; all Xf reads done (Y may alias it now)

    // ---- Layer 1: 48x256 @ 256x256 ---------------------------------------
    #pragma unroll
    for (int j = 0; j < 6; ++j) acc[j] = vzero;
    #pragma unroll
    for (int ks = 0; ks < 8; ++ks) {
        v16h a0 = *(const v16h*)(Hf + (0 * 8 + ks) * 512 + laneOff);
        v16h a1 = *(const v16h*)(Hf + (1 * 8 + ks) * 512 + laneOff);
        v16h a2 = *(const v16h*)(Hf + (2 * 8 + ks) * 512 + laneOff);
        #pragma unroll
        for (int nn = 0; nn < 2; ++nn) {
            int n = 2 * wave + nn;
            v16h bf = *(const v16h*)(wp1 + (((ks << 4) + n) << 9) + laneOff);
            acc[nn * 3 + 0] = __builtin_amdgcn_wmma_f32_16x16x32_f16(
                false, a0, false, bf, (short)0, acc[nn * 3 + 0], false, false);
            acc[nn * 3 + 1] = __builtin_amdgcn_wmma_f32_16x16x32_f16(
                false, a1, false, bf, (short)0, acc[nn * 3 + 1], false, false);
            acc[nn * 3 + 2] = __builtin_amdgcn_wmma_f32_16x16x32_f16(
                false, a2, false, bf, (short)0, acc[nn * 3 + 2], false, false);
        }
    }

    // ---- Epilogue 1: BN + ReLU -> Y (f32, aliases dead Xf) ---------------
    {
        int ln = lane & 15, lh = lane >> 4;
        #pragma unroll
        for (int nn = 0; nn < 2; ++nn) {
            int n   = 2 * wave + nn;
            int col = n * 16 + ln;
            float sc = s1[col], bs = t1[col];
            #pragma unroll
            for (int m = 0; m < 3; ++m) {
                #pragma unroll
                for (int r = 0; r < 8; ++r) {
                    float y = fmaxf(acc[nn * 3 + m][r] * sc + bs, 0.0f);
                    int row = m * 16 + r + 8 * lh;
                    Y[row * 256 + col] = y;
                }
            }
        }
    }
    __syncthreads();

    // ---- k=3 reduction -> output (16 queries x 256 channels) -------------
    for (int e = tid; e < 16 * 256; e += 256) {
        int qr = e >> 8;
        int c  = e & 255;
        float s = Y[(qr * 3 + 0) * 256 + c]
                + Y[(qr * 3 + 1) * 256 + c]
                + Y[(qr * 3 + 2) * 256 + c];
        out[(size_t)(rowBase / 3 + qr) * 256 + c] = s;
    }
}

// ---------------------------------------------------------------------------
// Host launcher
// ---------------------------------------------------------------------------
extern "C" void kernel_launch(void* const* d_in, const int* in_sizes, int n_in,
                              void* d_out, int out_size, void* d_ws, size_t ws_size,
                              hipStream_t stream) {
    const float* xyz1    = (const float*)d_in[0];
    const float* xyz2    = (const float*)d_in[1];
    const float* points1 = (const float*)d_in[2];
    const float* feat2   = (const float*)d_in[3];
    const float* dir1    = (const float*)d_in[4];
    // d_in[5] = dir2 : dead in the reference (angle2 overwritten) — unused.
    const float* W0 = (const float*)d_in[6];
    const float* g0 = (const float*)d_in[7];
    const float* b0 = (const float*)d_in[8];
    const float* m0 = (const float*)d_in[9];
    const float* v0 = (const float*)d_in[10];
    const float* W1 = (const float*)d_in[11];
    const float* g1 = (const float*)d_in[12];
    const float* b1 = (const float*)d_in[13];
    const float* m1 = (const float*)d_in[14];
    const float* v1 = (const float*)d_in[15];
    float* out = (float*)d_out;

    char* ws = (char*)d_ws;
    int*       idx = (int*)      (ws + WS_IDX);
    float*     s0  = (float*)    (ws + WS_S0);
    float*     t0  = (float*)    (ws + WS_T0);
    float*     s1  = (float*)    (ws + WS_S1);
    float*     t1  = (float*)    (ws + WS_T1);
    _Float16*  wp0 = (_Float16*) (ws + WS_WP0);
    _Float16*  wp1 = (_Float16*) (ws + WS_WP1);

    (void)in_sizes; (void)n_in; (void)out_size; (void)ws_size;

    hipFuncSetAttribute((const void*)fused_mlp_kernel,
                        hipFuncAttributeMaxDynamicSharedMemorySize, L_TOTAL);

    bn_fold_kernel<<<1, 256, 0, stream>>>(g0, b0, m0, v0, g1, b1, m1, v1,
                                          s0, t0, s1, t1);
    pack_w_kernel<<<KPAD, 256, 0, stream>>>(W0, wp0, INCH);
    pack_w_kernel<<<256,  256, 0, stream>>>(W1, wp1, 256);
    knn_kernel<<<NQ / 256, 256, 0, stream>>>(xyz1, xyz2, idx);
    fused_mlp_kernel<<<NTILES, 256, L_TOTAL, stream>>>(
        idx, points1, xyz2, dir1, feat2, wp0, wp1, s0, t0, s1, t1, out);
}
